// RNNEnc_3564822856207
// MI455X (gfx1250) — matrix operand order repaired and checked by reading, not compiled, not run on hardware
//
#include <hip/hip_runtime.h>

// ---------------------------------------------------------------------------
// Bidirectional GRU encoder for MI455X (gfx1250, wave32, WMMA).
//   B=32, T=512, D=744.  K padded to 768 (24 k-tiles of 32),
//   per-gate N padded to 768 (16 WGs x 48 dims).
// Strategy:
//   * bf16 WMMA (v_wmma_f32_16x16x32_bf16) with f32 accumulation.
//   * gi = x @ W_ih^T precomputed full-chip-parallel when workspace allows
//     (halves the WMMA work on the serial recurrence critical path);
//     otherwise fused into the recurrence (template<bool USE_GI>).
//   * 16 workgroups per direction; each owns 48 hidden dims -> 144 gate cols.
//   * W_hh slice (216 KB) + h stage (48 KB) resident in CDNA5's 320 KB LDS.
//   * h staged per step via GLOBAL_LOAD_ASYNC_TO_LDS (ASYNCcnt) when available.
//   * per-step cross-WG sync via monotonic global atomic barrier.
// ---------------------------------------------------------------------------

#define D_IN    744
#define DPAD    768
#define BATCH   32
#define TLEN    512
#define NWG     16          // workgroups per direction
#define DSLICE  48          // hidden dims per WG
#define WAVES   9           // one wave per 16-col N-tile (3 gates x 3)
#define THREADS (WAVES * 32)
#define KTILES  (DPAD / 32) // 24
#define OUTD    (2 * D_IN)  // 1488
#define NGATE3  (3 * DPAD)  // 2304 padded gate columns

#if defined(__has_builtin)
#  if __has_builtin(__builtin_amdgcn_global_load_async_to_lds_b128) && \
      __has_builtin(__builtin_amdgcn_s_wait_asynccnt)
#    define HAVE_ASYNC_LDS 1
#  endif
#endif
#ifndef HAVE_ASYNC_LDS
#  define HAVE_ASYNC_LDS 0
#endif

typedef __bf16 bf16x16 __attribute__((ext_vector_type(16)));
typedef float  f32x8   __attribute__((ext_vector_type(8)));
// Builtin signature (from hipcc diagnostic): param 1 is
//   'int __vector_size__(4*sizeof(int)) __device__ *'  (v4i in addrspace(1));
// param 2 is the LDS-side v4i pointer (addrspace(3)).
typedef __attribute__((__vector_size__(4 * sizeof(int)))) int v4i_t;
typedef __attribute__((address_space(1))) v4i_t as1_v4i;
typedef __attribute__((address_space(3))) v4i_t as3_v4i;

static __device__ __forceinline__ unsigned short f2bf(float f) {
  unsigned u = __float_as_uint(f);
  u += 0x7FFFu + ((u >> 16) & 1u);      // round-to-nearest-even
  return (unsigned short)(u >> 16);
}

// 16B global -> LDS copy: async DMA (ASYNCcnt) when toolchain exposes it.
static __device__ __forceinline__ void copy16_g2l(const void* g, void* l) {
#if HAVE_ASYNC_LDS
  __builtin_amdgcn_global_load_async_to_lds_b128((as1_v4i*)g, (as3_v4i*)l, 0, 0);
#else
  *(uint4*)l = *(const uint4*)g;
#endif
}
static __device__ __forceinline__ void copy_g2l_wait() {
#if HAVE_ASYNC_LDS
  __builtin_amdgcn_s_wait_asynccnt(0);
#endif
}

// A-matrix fragment, 16x32 bf16, row-major source with given row stride.
// lanes 0-15 row M=r hold K[k0..k0+15]; lanes 16-31 hold K[k0+16..k0+31]
// (contiguous, matches load_b_frag).  Two 16-byte loads per lane.
static __device__ __forceinline__ bf16x16
load_a_frag(const __bf16* base, size_t row_stride, int mtile, int k0, int lane) {
  const int r  = lane & 15;
  const int hi = (lane >> 4) * 16;
  const __bf16* p = base + (size_t)(mtile * 16 + r) * row_stride + k0 + hi;
  union { bf16x16 v; uint4 q[2]; } u;
  u.q[0] = *(const uint4*)(p);        // elements 0..7
  u.q[1] = *(const uint4*)(p + 8);    // elements 8..15
  return u.v;
}

// B-matrix fragment, 32x16 bf16, B[k][n] = W[n][k] with W row-major (ldb cols).
// lanes 0-15: col N=r, K[k0..k0+15]; lanes 16-31: K[k0+16..k0+31].
// One contiguous 32-byte load per lane.  Works for global or LDS pointers.
static __device__ __forceinline__ bf16x16
load_b_frag(const __bf16* wbase, int ldb, int k0, int lane) {
  const int r  = lane & 15;
  const int hi = (lane >> 4) * 16;
  return *(const bf16x16*)(wbase + (size_t)r * ldb + k0 + hi);
}

static __device__ __forceinline__ float sigmoidf_(float x) {
  return 1.0f / (1.0f + __expf(-x));
}

// ---------------------------------------------------------------------------
// Prep kernel: fp32 -> padded bf16 weights / inputs, zero h state + barriers.
// Runs every launch (workspace is poisoned; graph replay must be deterministic).
// Padded weight layout: Wb[mat][gate][dp<768][k<768], mats: ih_f, hh_f, ih_b, hh_b
// ---------------------------------------------------------------------------
__global__ void gru_prep_kernel(const float* __restrict__ v_in,
                                const float* __restrict__ Wihf,
                                const float* __restrict__ Whhf,
                                const float* __restrict__ Wihb,
                                const float* __restrict__ Whhb,
                                unsigned short* __restrict__ Wb,
                                unsigned short* __restrict__ xb,
                                unsigned short* __restrict__ hb16,
                                float* __restrict__ hf32,
                                unsigned* __restrict__ ctrs) {
  const size_t idx0   = (size_t)blockIdx.x * blockDim.x + threadIdx.x;
  const size_t stride = (size_t)gridDim.x * blockDim.x;
  if (idx0 == 0) { ctrs[0] = 0u; ctrs[1] = 0u; }

  const float* srcs[4] = {Wihf, Whhf, Wihb, Whhb};
  const size_t WELE = (size_t)4 * 3 * DPAD * DPAD;
  for (size_t i = idx0; i < WELE; i += stride) {
    const size_t k  = i % DPAD;
    const size_t rp = i / DPAD;
    const size_t dp = rp % DPAD;
    const size_t gm = rp / DPAD;
    const size_t g  = gm % 3, m = gm / 3;
    float f = 0.0f;
    if (dp < D_IN && k < D_IN) f = srcs[m][(g * D_IN + dp) * D_IN + k];
    Wb[i] = f2bf(f);
  }
  const size_t XELE = (size_t)BATCH * TLEN * DPAD;
  for (size_t i = idx0; i < XELE; i += stride) {
    const size_t k  = i % DPAD;
    const size_t bt = i / DPAD;
    const float  f  = (k < D_IN) ? v_in[bt * D_IN + k] : 0.0f;
    xb[i] = f2bf(f);
  }
  const size_t HELE = (size_t)2 * 2 * BATCH * DPAD;   // [dir][buf][b][k]
  for (size_t i = idx0; i < HELE; i += stride) { hb16[i] = 0; hf32[i] = 0.0f; }
}

// ---------------------------------------------------------------------------
// gi precompute:  gi[dir][t][b][n] = x[b,t,:] . W_ih[dir][n,:]   (n in [0,2304))
// Fully parallel over (t, dir, n-tiles): runs at full-chip WMMA rate, no LDS.
// grid = (T=512, 18 ntile-groups, 2 dirs), block = 256 (8 waves, 1 ntile each).
// ---------------------------------------------------------------------------
__global__ void __launch_bounds__(256)
gi_gemm_kernel(const unsigned short* __restrict__ Wb,
               const unsigned short* __restrict__ xb,
               float* __restrict__ gi) {
  const int t    = blockIdx.x;
  const int grp  = blockIdx.y;
  const int dir  = blockIdx.z;
  const int wave = threadIdx.x >> 5;
  const int lane = threadIdx.x & 31;
  const int n0   = (grp * 8 + wave) * 16;        // 0..2288

  const __bf16* Wih  = (const __bf16*)Wb + (size_t)(dir ? 2 : 0) * 3 * DPAD * DPAD;
  const __bf16* Wrow = Wih + (size_t)n0 * DPAD;  // rows n0..n0+15 (gate-major padded)
  const __bf16* Xt   = (const __bf16*)xb + (size_t)t * DPAD;

  f32x8 a0 = {}, a1 = {};
  #pragma unroll 4
  for (int kt = 0; kt < KTILES; ++kt) {
    const int k0 = kt * 32;
    bf16x16 ax0 = load_a_frag(Xt, (size_t)TLEN * DPAD, 0, k0, lane);
    bf16x16 ax1 = load_a_frag(Xt, (size_t)TLEN * DPAD, 1, k0, lane);
    bf16x16 bw  = load_b_frag(Wrow, DPAD, k0, lane);
    a0 = __builtin_amdgcn_wmma_f32_16x16x32_bf16(false, ax0, false, bw, (short)0, a0, false, false);
    a1 = __builtin_amdgcn_wmma_f32_16x16x32_bf16(false, ax1, false, bw, (short)0, a1, false, false);
  }
  float* dst = gi + (size_t)(dir * TLEN + t) * BATCH * NGATE3;
  const int bhi = (lane >> 4) * 8;
  const int col = n0 + (lane & 15);
  #pragma unroll
  for (int v = 0; v < 8; ++v) {
    dst[(size_t)(bhi + v) * NGATE3 + col]      = a0[v];
    dst[(size_t)(16 + bhi + v) * NGATE3 + col] = a1[v];
  }
}

// ---------------------------------------------------------------------------
// Recurrence kernel.  grid = (16 WGs, 2 directions), block = 288 (9 waves).
// Dynamic LDS: W_hh slice [3][48][768] bf16 (216 KB)
//            + gi/gh staging 2x[3][32][48] f32 (36 KB)
//            + h stage [32][768] bf16 (48 KB)            -> 300 KB total.
// USE_GI=true: gi read from precomputed buffer (half the WMMAs per step).
// ---------------------------------------------------------------------------
template <bool USE_GI>
__global__ void __launch_bounds__(THREADS)
gru_recurrence_kernel(const unsigned short* __restrict__ Wb,
                      const unsigned short* __restrict__ xb,
                      const float* __restrict__ gi,
                      unsigned short* __restrict__ hb16,
                      float* __restrict__ hf32,
                      const float* __restrict__ v_in,
                      const float* __restrict__ bihf, const float* __restrict__ bhhf,
                      const float* __restrict__ bihb, const float* __restrict__ bhhb,
                      const int* __restrict__ cl_ptr,
                      float* __restrict__ out,
                      unsigned* __restrict__ ctrs) {
  const int wg   = blockIdx.x;       // 0..15   hidden-dim slice
  const int dir  = blockIdx.y;       // 0 fwd, 1 bwd
  const int tid  = threadIdx.x;
  const int wave = tid >> 5;
  const int lane = tid & 31;
  const int d0   = wg * DSLICE;

  extern __shared__ __align__(32) char smem[];
  __bf16* sWhh = (__bf16*)smem;                               // [144][768] bf16
  float*  sGI  = (float*)(smem + (size_t)3 * DSLICE * DPAD * 2);
  float*  sGH  = sGI + 3 * BATCH * DSLICE;                    // [3][32][48] each
  __bf16* sH   = (__bf16*)(sGH + 3 * BATCH * DSLICE);         // [32][768] bf16

  const __bf16* Wih = (const __bf16*)Wb + (size_t)(dir ? 2 : 0) * 3 * DPAD * DPAD;
  const __bf16* Whh = (const __bf16*)Wb + (size_t)(dir ? 3 : 1) * 3 * DPAD * DPAD;
  const float*  bi  = dir ? bihb : bihf;
  const float*  bh  = dir ? bhhb : bhhf;

  // --- Stage W_hh slice into LDS: local row g*48+j  <- global row g*768+d0+j
  // (async DMA when available; ASYNCcnt tracked, one wait per wave)
  for (int idx = tid; idx < 3 * DSLICE * (DPAD / 8); idx += THREADS) {
    const int row = idx / (DPAD / 8);
    const int c8  = idx % (DPAD / 8);
    const int g   = row / DSLICE, j = row % DSLICE;
    const __bf16* src = Whh + ((size_t)(g * DPAD + d0 + j)) * DPAD + c8 * 8;
    copy16_g2l(src, sWhh + (size_t)row * DPAD + c8 * 8);
  }
  copy_g2l_wait();
  __syncthreads();

  const int cl   = *cl_ptr;
  const int Tout = TLEN - 2 * cl;
  unsigned* ctr  = ctrs + dir;

  // wave -> one 16-column N-tile of one gate
  const int g      = wave / 3;
  const int nsub   = wave % 3;
  const __bf16* WihRow  = Wih + (size_t)(g * DPAD + d0 + nsub * 16) * DPAD;
  const __bf16* sWhhRow = sWhh + (size_t)(g * DSLICE + nsub * 16) * DPAD;

  const __bf16* xbp = (const __bf16*)xb;
  int cur = 0;

  for (int step = 0; step < TLEN; ++step) {
    const int t = dir ? (TLEN - 1 - step) : step;
    const __bf16* Xt = xbp + (size_t)t * DPAD;                       // rows: b, stride T*DPAD
    const __bf16* Hc = (const __bf16*)hb16 + (size_t)(dir * 2 + cur) * BATCH * DPAD;

    // --- stage h(t) once into LDS (all 9 waves would otherwise re-read it 9x)
    for (int idx = tid; idx < BATCH * DPAD / 8; idx += THREADS)
      copy16_g2l(Hc + (size_t)idx * 8, sH + (size_t)idx * 8);
    copy_g2l_wait();
    __syncthreads();

    f32x8 accI0 = {}, accI1 = {}, accH0 = {}, accH1 = {};
    #pragma unroll 4
    for (int kt = 0; kt < KTILES; ++kt) {
      const int k0 = kt * 32;
      bf16x16 ah0 = load_a_frag(sH, DPAD, 0, k0, lane);
      bf16x16 ah1 = load_a_frag(sH, DPAD, 1, k0, lane);
      bf16x16 bfh = load_b_frag(sWhhRow, DPAD, k0, lane);
      accH0 = __builtin_amdgcn_wmma_f32_16x16x32_bf16(false, ah0, false, bfh, (short)0, accH0, false, false);
      accH1 = __builtin_amdgcn_wmma_f32_16x16x32_bf16(false, ah1, false, bfh, (short)0, accH1, false, false);
      if (!USE_GI) {
        bf16x16 ax0 = load_a_frag(Xt, (size_t)TLEN * DPAD, 0, k0, lane);
        bf16x16 ax1 = load_a_frag(Xt, (size_t)TLEN * DPAD, 1, k0, lane);
        bf16x16 bfi = load_b_frag(WihRow, DPAD, k0, lane);
        if (kt + 1 < KTILES)   // pull next W_ih tile into WGP-level cache
          __builtin_prefetch(WihRow + (size_t)(lane & 15) * DPAD + k0 + 32, 0, 3);
        accI0 = __builtin_amdgcn_wmma_f32_16x16x32_bf16(false, ax0, false, bfi, (short)0, accI0, false, false);
        accI1 = __builtin_amdgcn_wmma_f32_16x16x32_bf16(false, ax1, false, bfi, (short)0, accI1, false, false);
      }
    }

    // C layout (ISA 7.12.2): lane<16 -> (M=v, N=lane); lane>=16 -> (M=8+v, N=lane-16)
    {
      const int bhi = (lane >> 4) * 8;
      const int jl  = nsub * 16 + (lane & 15);
      #pragma unroll
      for (int v = 0; v < 8; ++v) {
        sGH[(g * BATCH + bhi + v)      * DSLICE + jl] = accH0[v];
        sGH[(g * BATCH + 16 + bhi + v) * DSLICE + jl] = accH1[v];
        if (!USE_GI) {
          sGI[(g * BATCH + bhi + v)      * DSLICE + jl] = accI0[v];
          sGI[(g * BATCH + 16 + bhi + v) * DSLICE + jl] = accI1[v];
        }
      }
    }
    __syncthreads();

    // --- elementwise GRU cell on this WG's 32x48 slice
    const float* giRow = USE_GI ? (gi + (size_t)(dir * TLEN + t) * BATCH * NGATE3) : nullptr;
    for (int idx = tid; idx < BATCH * DSLICE; idx += THREADS) {
      const int b = idx / DSLICE, j = idx % DSLICE;
      const int d = d0 + j;
      if (d < D_IN) {
        float gir, giz, gin;
        if (USE_GI) {
          const float* gb = giRow + (size_t)b * NGATE3 + d;
          gir = gb[0 * DPAD]; giz = gb[1 * DPAD]; gin = gb[2 * DPAD];
        } else {
          gir = sGI[(0 * BATCH + b) * DSLICE + j];
          giz = sGI[(1 * BATCH + b) * DSLICE + j];
          gin = sGI[(2 * BATCH + b) * DSLICE + j];
        }
        gir += bi[0 * D_IN + d];
        giz += bi[1 * D_IN + d];
        gin += bi[2 * D_IN + d];
        const float ghr = sGH[(0 * BATCH + b) * DSLICE + j] + bh[0 * D_IN + d];
        const float ghz = sGH[(1 * BATCH + b) * DSLICE + j] + bh[1 * D_IN + d];
        const float ghn = sGH[(2 * BATCH + b) * DSLICE + j] + bh[2 * D_IN + d];
        const float r = sigmoidf_(gir + ghr);
        const float z = sigmoidf_(giz + ghz);
        const float n = tanhf(gin + r * ghn);
        const float hp = hf32[((size_t)(dir * 2 + cur) * BATCH + b) * DPAD + d];
        const float hn = (1.0f - z) * n + z * hp;
        hf32[((size_t)(dir * 2 + (cur ^ 1)) * BATCH + b) * DPAD + d] = hn;
        hb16[((size_t)(dir * 2 + (cur ^ 1)) * BATCH + b) * DPAD + d] = f2bf(hn);
        if (step >= cl && step < TLEN - cl) {
          const float x = v_in[((size_t)b * TLEN + t) * D_IN + d];
          out[((size_t)b * Tout + (step - cl)) * OUTD + dir * D_IN + d] = hn + x;
        }
      }
    }

    // --- device-scope barrier among the 16 WGs of this direction.
    // Monotonic counter: after barrier #n (1-based) counter == n*NWG.
    __threadfence();
    __syncthreads();
    if (tid == 0) {
      __hip_atomic_fetch_add(ctr, 1u, __ATOMIC_RELEASE, __HIP_MEMORY_SCOPE_AGENT);
      const unsigned target = (unsigned)(step + 1) * (unsigned)NWG;
      while (__hip_atomic_load(ctr, __ATOMIC_ACQUIRE, __HIP_MEMORY_SCOPE_AGENT) < target)
        __builtin_amdgcn_s_sleep(2);
    }
    __syncthreads();
    cur ^= 1;
  }
}

// ---------------------------------------------------------------------------
extern "C" void kernel_launch(void* const* d_in, const int* in_sizes, int n_in,
                              void* d_out, int out_size, void* d_ws, size_t ws_size,
                              hipStream_t stream) {
  (void)in_sizes; (void)n_in; (void)out_size;
  const float* v_in  = (const float*)d_in[0];
  const float* Wihf  = (const float*)d_in[1];
  const float* Whhf  = (const float*)d_in[2];
  const float* bihf  = (const float*)d_in[3];
  const float* bhhf  = (const float*)d_in[4];
  const float* Wihb  = (const float*)d_in[5];
  const float* Whhb  = (const float*)d_in[6];
  const float* bihb  = (const float*)d_in[7];
  const float* bhhb  = (const float*)d_in[8];
  const int*   clp   = (const int*)d_in[9];
  float*       out   = (float*)d_out;

  unsigned char* ws = (unsigned char*)d_ws;
  size_t off = 0;
  unsigned* ctrs = (unsigned*)(ws + off);            off += 256;
  unsigned short* Wb = (unsigned short*)(ws + off);  off += (size_t)4 * 3 * DPAD * DPAD * 2; // 14.2 MB
  unsigned short* xb = (unsigned short*)(ws + off);  off += (size_t)BATCH * TLEN * DPAD * 2; // 25.2 MB
  unsigned short* hb16 = (unsigned short*)(ws + off); off += (size_t)2 * 2 * BATCH * DPAD * 2;
  float* hf32 = (float*)(ws + off);                  off += (size_t)2 * 2 * BATCH * DPAD * 4;
  const size_t gi_bytes = (size_t)2 * TLEN * BATCH * NGATE3 * 4;  // 302 MB
  const bool use_gi = (ws_size >= off + gi_bytes);
  float* gi = use_gi ? (float*)(ws + off) : nullptr;

  gru_prep_kernel<<<2048, 256, 0, stream>>>(v_in, Wihf, Whhf, Wihb, Whhb,
                                            Wb, xb, hb16, hf32, ctrs);

  const size_t smem_bytes = (size_t)3 * DSLICE * DPAD * 2          // W_hh slice (216 KB)
                          + (size_t)2 * 3 * BATCH * DSLICE * 4     // gi/gh staging (36 KB)
                          + (size_t)BATCH * DPAD * 2;              // h stage (48 KB)
  if (use_gi) {
    gi_gemm_kernel<<<dim3(TLEN, 18, 2), 256, 0, stream>>>(Wb, xb, gi);
    gru_recurrence_kernel<true><<<dim3(NWG, 2), THREADS, smem_bytes, stream>>>(
        Wb, xb, gi, hb16, hf32, v_in, bihf, bhhf, bihb, bhhb, clp, out, ctrs);
  } else {
    gru_recurrence_kernel<false><<<dim3(NWG, 2), THREADS, smem_bytes, stream>>>(
        Wb, xb, nullptr, hb16, hf32, v_in, bihf, bhhf, bihb, bhhb, clp, out, ctrs);
  }
}
